// PointTransformerGNN_6004364280126
// MI455X (gfx1250) — matrix operand order repaired and checked by reading, not compiled, not run on hardware
//
#include <hip/hip_runtime.h>

#define Hdim 256
#define Ldim 4
#define NHdim 8
#define Sdim 2
#define Ndim 1024
#define Edim 32768

typedef __attribute__((ext_vector_type(16))) __bf16         v16bf;
typedef __attribute__((ext_vector_type(16))) unsigned short v16us;
typedef __attribute__((ext_vector_type(8)))  unsigned short v8us;
typedef __attribute__((ext_vector_type(8)))  float          v8f;

// ---------------------------------------------------------------------------
// float -> bf16 (round-to-nearest-even), via bit manipulation
// ---------------------------------------------------------------------------
static __device__ __forceinline__ unsigned short f2bf(float f) {
  unsigned u = __builtin_bit_cast(unsigned, f);
  u += 0x7fffu + ((u >> 16) & 1u);
  return (unsigned short)(u >> 16);
}

// ---------------------------------------------------------------------------
// A fragment for v_wmma_f32_16x16x32_bf16 from a bf16 row (per-lane pointer
// ap already includes row*lda + (lane>=16 ? 8 : 0)).  Two aligned 16B loads.
// elems 0-7 = K k0+kb..+7, elems 8-15 = K k0+kb+16..+23  (kb=0|8 per lane).
// ---------------------------------------------------------------------------
static __device__ __forceinline__ v16bf load_a_frag_bf(
    const unsigned short* __restrict__ ap, int k0) {
  const v8us lo = *(const v8us*)(ap + k0);
  const v8us hi = *(const v8us*)(ap + k0 + 16);
  const v16us u = __builtin_shufflevector(lo, hi, 0, 1, 2, 3, 4, 5, 6, 7, 8, 9,
                                          10, 11, 12, 13, 14, 15);
  return __builtin_bit_cast(v16bf, u);
}

// ---------------------------------------------------------------------------
// B fragment from a pre-converted bf16 weight matrix Wb (N x ldw, row-major),
// C = A@W^T.  Two aligned 16-byte vector loads.
// N=n0+lane%16; elems 0-7 = K k0+kb..+7, elems 8-15 = K k0+kb+16..+23 (kb=0|8).
// ---------------------------------------------------------------------------
static __device__ __forceinline__ v16bf load_b_frag_bf(
    const unsigned short* __restrict__ Wb, int ldw, int n0, int k0) {
  const int lane = threadIdx.x & 31;
  const unsigned short* p =
      Wb + (long)(n0 + (lane & 15)) * ldw + k0 + ((lane >> 4) << 3);
  const v8us lo = *(const v8us*)p;
  const v8us hi = *(const v8us*)(p + 16);
  const v16us u = __builtin_shufflevector(lo, hi, 0, 1, 2, 3, 4, 5, 6, 7, 8, 9,
                                          10, 11, 12, 13, 14, 15);
  return __builtin_bit_cast(v16bf, u);
}

// ---------------------------------------------------------------------------
// Cooperative staging of a 256(N) x 32(K) bf16 weight slice into LDS.
// Row stride 40 ushorts (80B): 16B-aligned vector accesses, and the 16 rows
// of one fragment map to distinct LDS bank groups.  128 threads, 2 rows each.
// ---------------------------------------------------------------------------
#define BSL_STRIDE 40
static __device__ __forceinline__ void stage_b_slice(
    const unsigned short* __restrict__ W, int ldw, int k0,
    unsigned short (*bsl)[BSL_STRIDE], int tid) {
  const int row0 = tid << 1;
#pragma unroll
  for (int rr = 0; rr < 2; ++rr) {
    const unsigned short* s = W + (long)(row0 + rr) * ldw + k0;
    *(v8us*)&bsl[row0 + rr][0]  = *(const v8us*)(s);
    *(v8us*)&bsl[row0 + rr][8]  = *(const v8us*)(s + 8);
    *(v8us*)&bsl[row0 + rr][16] = *(const v8us*)(s + 16);
    *(v8us*)&bsl[row0 + rr][24] = *(const v8us*)(s + 24);
  }
}

static __device__ __forceinline__ v16bf b_frag_from_lds(
    const unsigned short (*bsl)[BSL_STRIDE], int n0) {
  const int lane = threadIdx.x & 31;
  const unsigned short* p = &bsl[n0 + (lane & 15)][(lane >> 4) << 3];
  const v8us lo = *(const v8us*)p;
  const v8us hi = *(const v8us*)(p + 16);
  const v16us u = __builtin_shufflevector(lo, hi, 0, 1, 2, 3, 4, 5, 6, 7, 8, 9,
                                          10, 11, 12, 13, 14, 15);
  return __builtin_bit_cast(v16bf, u);
}

// ---------------------------------------------------------------------------
// Generic GEMM:  C[M,N] = act(A[M,K] @ W[N,K]^T + bias (+ resid))
// A and W are bf16 (produced by upstream kernels); accumulate fp32.
// block = 128 threads (4 waves); block tile 64(M) x 64(N); each wave owns a
// 16-row stripe with 4 accumulators.  Requires M%16==0, N%64==0, K%32==0.
// Template flags keep the epilogue straight-line (no per-element branches).
// ---------------------------------------------------------------------------
template <int RELU, int RESID, int WB>
__global__ __launch_bounds__(128) void gemm_bias_kernel(
    const unsigned short* __restrict__ Ab,
    const unsigned short* __restrict__ Wb, const float* __restrict__ bias,
    const float* __restrict__ resid, float* __restrict__ C,
    unsigned short* __restrict__ Cbf, int M, int N, int K, int lda, int ldw,
    int ldc) {
  const int w = threadIdx.x >> 5;
  const int lane = threadIdx.x & 31;
  const int m0 = blockIdx.x * 64 + w * 16;
  const int n0 = blockIdx.y * 64;
  if (m0 >= M) return;

  v8f z = {};
  v8f acc[4];
#pragma unroll
  for (int j = 0; j < 4; ++j) acc[j] = z;

  const unsigned short* ap =
      Ab + (long)(m0 + (lane & 15)) * lda + ((lane >> 4) << 3);
  for (int k0 = 0; k0 < K; k0 += 32) {
    __builtin_prefetch(ap + k0 + 128, 0, 1);  // global_prefetch_b8
    const v16bf af = load_a_frag_bf(ap, k0);
#pragma unroll
    for (int j = 0; j < 4; ++j) {
      const v16bf bfr = load_b_frag_bf(Wb, ldw, n0 + 16 * j, k0);
      acc[j] = __builtin_amdgcn_wmma_f32_16x16x32_bf16(
          false, af, false, bfr, (short)0, acc[j], false, false);
    }
  }

  // C/D layout: lanes 0-15 -> N=lane, rows m0..m0+7; lanes 16-31 -> rows +8.
  const int mb = m0 + ((lane >> 4) << 3);
#pragma unroll
  for (int j = 0; j < 4; ++j) {
    const int nn = n0 + 16 * j + (lane & 15);
    const float bv = bias[nn];
#pragma unroll
    for (int rr = 0; rr < 8; ++rr) {
      const long ci = (long)(mb + rr) * ldc + nn;
      float v = acc[j][rr] + bv;
      if (RESID) v += resid[ci];
      if (RELU) v = fmaxf(v, 0.f);
      C[ci] = v;
      if (WB) Cbf[ci] = f2bf(v);
    }
  }
}

// ---------------------------------------------------------------------------
// Fused edge MLP (the dominant cost):
//   mid = relu( concat(a[b,col[e],:], ea[b,e,:]) @ w1^T + b1 )   (K=259 -> 288)
//   msg = mid @ w2^T + b2
//   agg[b,row[e],:] += msg        (atomic scatter; bias per edge as in ref)
// Gather fused into the A-frag load; mid kept as bf16 in LDS.  The B weight
// slice (256x32) is staged cooperatively in LDS each K-step and shared by all
// 4 waves (kills the 4x redundant global B traffic); fragments come from
// ds_load_b128.  w1b is zero-padded to K=288 so only the peeled last K
// iteration (ea concat) has a conditional A path.
// ---------------------------------------------------------------------------
__global__ __launch_bounds__(128) void edge_mlp_kernel(
    const unsigned short* __restrict__ abf,
    const unsigned short* __restrict__ eabf, const int* __restrict__ rowi,
    const int* __restrict__ coli, const unsigned short* __restrict__ w1b,
    const float* __restrict__ b1, const unsigned short* __restrict__ w2b,
    const float* __restrict__ b2, float* __restrict__ agg) {
  __shared__ alignas(64) unsigned short mid[4][16][256];       // 32 KB
  __shared__ alignas(64) unsigned short bsl[256][BSL_STRIDE];  // 20 KB
  const int tid = threadIdx.x;
  const int w = tid >> 5;
  const int lane = tid & 31;
  const int m0 = blockIdx.x * 64 + w * 16;  // flattened edge row (b*E+e)
  const int r = m0 + (lane & 15);
  const int b = r / Edim;
  const int e = r - b * Edim;
  const int kb_a = (lane >> 4) << 3;
  const unsigned short* ap = abf + ((long)(b * Ndim + coli[e])) * Hdim + kb_a;
  const long ebase = (long)r * 3;

  v8f z = {};
  v8f acc[16];
#pragma unroll
  for (int j = 0; j < 16; ++j) acc[j] = z;

  // ---- stage 1 main loop: K = 0..255, fully unconditional ----
  for (int k0 = 0; k0 < 256; k0 += 32) {
    stage_b_slice(w1b, Hdim + 32, k0, bsl, tid);
    __builtin_prefetch(ap + k0 + 128, 0, 1);
    const v16bf af = load_a_frag_bf(ap, k0);
    __syncthreads();
#pragma unroll
    for (int j = 0; j < 16; ++j) {
      const v16bf bfr = b_frag_from_lds(bsl, j * 16);
      acc[j] = __builtin_amdgcn_wmma_f32_16x16x32_bf16(
          false, af, false, bfr, (short)0, acc[j], false, false);
    }
    __syncthreads();
  }
  // ---- stage 1 peeled tail: K = 256..287 (ea concat + zero pad) ----
  {
    stage_b_slice(w1b, Hdim + 32, 256, bsl, tid);
    v16us ua;
#pragma unroll
    for (int i = 0; i < 16; ++i) {
      const int k = 256 + kb_a + ((i < 8) ? i : (i + 8));
      ua[i] = (k < 259) ? eabf[ebase + (k - 256)] : (unsigned short)0;
    }
    const v16bf af = __builtin_bit_cast(v16bf, ua);
    __syncthreads();
#pragma unroll
    for (int j = 0; j < 16; ++j) {
      const v16bf bfr = b_frag_from_lds(bsl, j * 16);
      acc[j] = __builtin_amdgcn_wmma_f32_16x16x32_bf16(
          false, af, false, bfr, (short)0, acc[j], false, false);
    }
    __syncthreads();
  }

  const int ncol = lane & 15;
  const int mbl = (lane >> 4) << 3;
#pragma unroll
  for (int j = 0; j < 16; ++j) {
    const int n = j * 16 + ncol;
    const float bv = b1[n];
#pragma unroll
    for (int rr = 0; rr < 8; ++rr)
      mid[w][mbl + rr][n] = f2bf(fmaxf(acc[j][rr] + bv, 0.f));
  }
  __syncthreads();

  // ---- stage 2: mid(16x256 bf16 in LDS) @ w2^T, K=256 ----
  v8f acc2[16];
#pragma unroll
  for (int j = 0; j < 16; ++j) acc2[j] = z;

  const unsigned short* mp = &mid[w][lane & 15][0] + kb_a;
  for (int k0 = 0; k0 < 256; k0 += 32) {
    stage_b_slice(w2b, Hdim, k0, bsl, tid);
    const v16bf af = load_a_frag_bf(mp, k0);
    __syncthreads();
#pragma unroll
    for (int j = 0; j < 16; ++j) {
      const v16bf bfr = b_frag_from_lds(bsl, j * 16);
      acc2[j] = __builtin_amdgcn_wmma_f32_16x16x32_bf16(
          false, af, false, bfr, (short)0, acc2[j], false, false);
    }
    __syncthreads();
  }

  // ---- scatter-aggregate epilogue ----
  int dst[8];
#pragma unroll
  for (int rr = 0; rr < 8; ++rr) {
    const int r2 = m0 + mbl + rr;
    const int b2i = r2 / Edim;
    const int e2 = r2 - b2i * Edim;
    dst[rr] = b2i * Ndim + rowi[e2];
  }
#pragma unroll
  for (int j = 0; j < 16; ++j) {
    const int n = j * 16 + ncol;
    const float bv = b2[n];
#pragma unroll
    for (int rr = 0; rr < 8; ++rr)
      atomicAdd(agg + (long)dst[rr] * Hdim + n, acc2[j][rr] + bv);
  }
}

// ---------------------------------------------------------------------------
// LayerNorm over H=256, one row per 256-thread block.
//   t = x (+resid) (*rscale[row%smod]);  y = LN(t)*g+b (+post); ybf optional
// ---------------------------------------------------------------------------
__global__ __launch_bounds__(256) void ln_kernel(
    const float* __restrict__ x, const float* __restrict__ resid,
    const float* __restrict__ rscale, int smod, const float* __restrict__ g,
    const float* __restrict__ bb, const float* __restrict__ post,
    float* __restrict__ y, unsigned short* __restrict__ ybf) {
  __shared__ float red[256];
  __shared__ float stat[2];
  const int rowi = blockIdx.x;
  const int t = threadIdx.x;
  const long off = (long)rowi * Hdim;
  float v = x[off + t];
  if (resid) v += resid[off + t];
  if (rscale) v *= rscale[rowi % smod];
  red[t] = v;
  __syncthreads();
  for (int s = 128; s > 0; s >>= 1) {
    if (t < s) red[t] += red[t + s];
    __syncthreads();
  }
  if (t == 0) stat[0] = red[0] * (1.f / 256.f);
  __syncthreads();
  const float m = stat[0];
  const float d = v - m;
  red[t] = d * d;
  __syncthreads();
  for (int s = 128; s > 0; s >>= 1) {
    if (t < s) red[t] += red[t + s];
    __syncthreads();
  }
  if (t == 0) stat[1] = red[0] * (1.f / 256.f);
  __syncthreads();
  const float rstd = rsqrtf(stat[1] + 1e-5f);
  float o = d * rstd * g[t] + bb[t];
  if (post) o += post[off + t];
  y[off + t] = o;
  if (ybf) ybf[off + t] = f2bf(o);
}

// ---------------------------------------------------------------------------
// Graph MHA core: B=2, NH=8, T=1024, d=32. One wave per query.
// qkv rows: (b*T+t)*768 ; q|k|v at col offsets 0|256|512, head*32.
// Output written as bf16 (consumed only as a GEMM A-matrix).
// ---------------------------------------------------------------------------
__global__ __launch_bounds__(256) void graph_attn_kernel(
    const float* __restrict__ qkv, unsigned short* __restrict__ out) {
  __shared__ float qs[8][32];
  __shared__ float sc[8][1024];
  const int w = threadIdx.x >> 5;
  const int lane = threadIdx.x & 31;
  const int wid = blockIdx.x * 8 + w;
  const int q = wid % Ndim;
  const int bh = wid / Ndim;
  const int hh = bh & 7;
  const int b = bh >> 3;
  const long rowq = (long)(b * Ndim + q);
  qs[w][lane] = qkv[rowq * 768 + hh * 32 + lane];
  __syncthreads();

  const float scl = 0.17677669529663687f;  // 1/sqrt(32)
  float mx = -1e30f;
  for (int j = 0; j < 32; ++j) {
    const int k = lane + (j << 5);
    const float* kp = qkv + (long)(b * Ndim + k) * 768 + 256 + hh * 32;
    float s = 0.f;
#pragma unroll
    for (int d = 0; d < 32; ++d) s += qs[w][d] * kp[d];
    s *= scl;
    sc[w][k] = s;
    mx = fmaxf(mx, s);
  }
#pragma unroll
  for (int o = 16; o > 0; o >>= 1) mx = fmaxf(mx, __shfl_xor(mx, o, 32));
  float sum = 0.f;
  for (int j = 0; j < 32; ++j) {
    const int k = lane + (j << 5);
    const float e = __expf(sc[w][k] - mx);
    sc[w][k] = e;
    sum += e;
  }
#pragma unroll
  for (int o = 16; o > 0; o >>= 1) sum += __shfl_xor(sum, o, 32);
  const float inv = 1.f / sum;
  __syncthreads();

  float o = 0.f;
  for (int k = 0; k < Ndim; ++k)
    o += sc[w][k] * qkv[(long)(b * Ndim + k) * 768 + 512 + hh * 32 + lane];
  out[rowq * Hdim + hh * 32 + lane] = f2bf(o * inv);
}

// ---------------------------------------------------------------------------
// Transformer MHA over T=2 (trivial): one thread per (node, head).
// ---------------------------------------------------------------------------
__global__ void t_attn_kernel(const float* __restrict__ qkv,
                              unsigned short* __restrict__ out) {
  const int i = blockIdx.x * blockDim.x + threadIdx.x;
  if (i >= Ndim * NHdim) return;
  const int hh = i & 7;
  const int n = i >> 3;
  const float* r0 = qkv + (long)(n * 2 + 0) * 768;
  const float* r1 = qkv + (long)(n * 2 + 1) * 768;
  const float* q0 = r0 + hh * 32, *q1 = r1 + hh * 32;
  const float* k0 = r0 + 256 + hh * 32, *k1 = r1 + 256 + hh * 32;
  const float* v0 = r0 + 512 + hh * 32, *v1 = r1 + 512 + hh * 32;
  float s00 = 0, s01 = 0, s10 = 0, s11 = 0;
#pragma unroll
  for (int d = 0; d < 32; ++d) {
    s00 += q0[d] * k0[d];
    s01 += q0[d] * k1[d];
    s10 += q1[d] * k0[d];
    s11 += q1[d] * k1[d];
  }
  const float scl = 0.17677669529663687f;
  s00 *= scl; s01 *= scl; s10 *= scl; s11 *= scl;
  const float m0 = fmaxf(s00, s01), m1 = fmaxf(s10, s11);
  const float e00 = __expf(s00 - m0), e01 = __expf(s01 - m0);
  const float e10 = __expf(s10 - m1), e11 = __expf(s11 - m1);
  const float i0 = 1.f / (e00 + e01), i1 = 1.f / (e10 + e11);
  const float p00 = e00 * i0, p01 = e01 * i0, p10 = e10 * i1, p11 = e11 * i1;
  unsigned short* o0 = out + (long)(n * 2 + 0) * Hdim + hh * 32;
  unsigned short* o1 = out + (long)(n * 2 + 1) * Hdim + hh * 32;
#pragma unroll
  for (int d = 0; d < 32; ++d) {
    o0[d] = f2bf(p00 * v0[d] + p01 * v1[d]);
    o1[d] = f2bf(p10 * v0[d] + p11 * v1[d]);
  }
}

// ---------------------------------------------------------------------------
// Small helper kernels
// ---------------------------------------------------------------------------
__global__ void f32_to_bf16_kernel(const float* __restrict__ src,
                                   unsigned short* __restrict__ dst, long n) {
  long i = (long)blockIdx.x * blockDim.x + threadIdx.x;
  const long stride = (long)gridDim.x * blockDim.x;
  for (; i < n; i += stride) dst[i] = f2bf(src[i]);
}

__global__ void f32_to_bf16_pad_kernel(const float* __restrict__ src,
                                       unsigned short* __restrict__ dst,
                                       int rows, int sld, int dld) {
  const int i = blockIdx.x * blockDim.x + threadIdx.x;
  if (i >= rows * dld) return;
  const int r = i / dld;
  const int c = i - r * dld;
  dst[i] = (c < sld) ? f2bf(src[(long)r * sld + c]) : (unsigned short)0;
}

// fe layer 1 (K=3): output consumed only as GEMM A -> write bf16 directly
__global__ void fe1_kernel(const float* __restrict__ x,
                           const float* __restrict__ w1,
                           const float* __restrict__ b1,
                           unsigned short* __restrict__ o, int rows) {
  const int i = blockIdx.x * blockDim.x + threadIdx.x;
  if (i >= rows * Hdim) return;
  const int r = i >> 8;
  const int j = i & 255;
  const float* xr = x + r * 3;
  const float* wj = w1 + j * 3;
  const float v = xr[0] * wj[0] + xr[1] * wj[1] + xr[2] * wj[2] + b1[j];
  o[i] = f2bf(fmaxf(v, 0.f));
}

// ea = pos[col] - pos[row]; consumed only by the edge tail -> bf16
__global__ void ea_kernel(const float* __restrict__ h,
                          const int* __restrict__ rowi,
                          const int* __restrict__ coli,
                          unsigned short* __restrict__ ea) {
  const int i = blockIdx.x * blockDim.x + threadIdx.x;
  if (i >= Sdim * Edim) return;
  const int b = i / Edim;
  const int e = i - b * Edim;
  const float* pc = h + (long)(b * Ndim + coli[e]) * Hdim;
  const float* pr = h + (long)(b * Ndim + rowi[e]) * Hdim;
  unsigned short* o = ea + (long)i * 3;
  o[0] = f2bf(pc[0] - pr[0]);
  o[1] = f2bf(pc[1] - pr[1]);
  o[2] = f2bf(pc[2] - pr[2]);
}

__global__ void zero_kernel(float* __restrict__ p, long n) {
  long i = (long)blockIdx.x * blockDim.x + threadIdx.x;
  const long stride = (long)gridDim.x * blockDim.x;
  for (; i < n; i += stride) p[i] = 0.f;
}

__global__ void count_kernel(const int* __restrict__ rowi,
                             float* __restrict__ cnt) {
  const int i = blockIdx.x * blockDim.x + threadIdx.x;
  if (i < Edim) atomicAdd(cnt + rowi[i], 1.0f);
}

__global__ void inv_kernel(float* __restrict__ cnt, int n) {
  const int i = blockIdx.x * blockDim.x + threadIdx.x;
  if (i < n) cnt[i] = 1.f / fmaxf(cnt[i], 1.f);
}

// (S,N,H) -> (N,S,H); also emit bf16 mirror (first transformer GEMM input)
__global__ void transpose_kernel(const float* __restrict__ h,
                                 float* __restrict__ ht,
                                 unsigned short* __restrict__ htbf) {
  const int i = blockIdx.x * blockDim.x + threadIdx.x;
  if (i >= Sdim * Ndim * Hdim) return;
  const int c = i & 255;
  const int rn = i >> 8;
  const int s = rn / Ndim;
  const int n = rn - s * Ndim;
  const float v = h[i];
  const long di = (long)(n * Sdim + s) * Hdim + c;
  ht[di] = v;
  htbf[di] = f2bf(v);
}

__global__ void out_kernel(const float* __restrict__ o1,
                           const float* __restrict__ w2,
                           const float* __restrict__ b2,
                           float* __restrict__ out) {
  const int i = blockIdx.x * blockDim.x + threadIdx.x;
  if (i >= Ndim * 3) return;
  const int n = i / 3;
  const int j = i - n * 3;
  const float* a = o1 + (long)n * Hdim;
  const float* w = w2 + (long)j * Hdim;
  float s = b2[j];
  for (int k = 0; k < Hdim; ++k) s += a[k] * w[k];
  out[i] = s;
}

// ---------------------------------------------------------------------------
// Launcher
// ---------------------------------------------------------------------------
extern "C" void kernel_launch(void* const* d_in, const int* in_sizes, int n_in,
                              void* d_out, int out_size, void* d_ws,
                              size_t ws_size, hipStream_t stream) {
  (void)in_sizes; (void)n_in; (void)out_size; (void)ws_size;
  const float* x      = (const float*)d_in[0];
  const int*   ei     = (const int*)  d_in[1];
  const float* fe_w1  = (const float*)d_in[2];
  const float* fe_b1  = (const float*)d_in[3];
  const float* fe_w2  = (const float*)d_in[4];
  const float* fe_b2  = (const float*)d_in[5];
  const float* g_inw  = (const float*)d_in[6];
  const float* g_inb  = (const float*)d_in[7];
  const float* g_outw = (const float*)d_in[8];
  const float* g_outb = (const float*)d_in[9];
  const float* g_m1w  = (const float*)d_in[10];
  const float* g_m1b  = (const float*)d_in[11];
  const float* g_m2w  = (const float*)d_in[12];
  const float* g_m2b  = (const float*)d_in[13];
  const float* g_n1g  = (const float*)d_in[14];
  const float* g_n1b  = (const float*)d_in[15];
  const float* g_n2g  = (const float*)d_in[16];
  const float* g_n2b  = (const float*)d_in[17];
  const float* t_inw  = (const float*)d_in[18];
  const float* t_inb  = (const float*)d_in[19];
  const float* t_outw = (const float*)d_in[20];
  const float* t_outb = (const float*)d_in[21];
  const float* t_l1w  = (const float*)d_in[22];
  const float* t_l1b  = (const float*)d_in[23];
  const float* t_l2w  = (const float*)d_in[24];
  const float* t_l2b  = (const float*)d_in[25];
  const float* t_n1g  = (const float*)d_in[26];
  const float* t_n1b  = (const float*)d_in[27];
  const float* t_n2g  = (const float*)d_in[28];
  const float* t_n2b  = (const float*)d_in[29];
  const float* o_w1   = (const float*)d_in[30];
  const float* o_b1   = (const float*)d_in[31];
  const float* o_w2   = (const float*)d_in[32];
  const float* o_b2   = (const float*)d_in[33];
  const int* rowi = ei;
  const int* coli = ei + Edim;

  float* ws = (float*)d_ws;
  size_t off = 0;
  auto alloc = [&](size_t n) { float* p = ws + off; off += n; return p; };
  const size_t R = (size_t)Sdim * Ndim;  // 2048 token rows
  float* h    = alloc(R * Hdim);
  float* hn   = alloc(R * Hdim);
  float* qkv  = alloc(R * 3 * Hdim);
  float* abuf = alloc(R * Hdim);
  float* cnt  = alloc(Ndim);
  float* agg  = alloc(R * Hdim);
  float* ht   = alloc(R * Hdim);
  float* ff1  = alloc(R * 4 * Hdim);
  float* ff2  = alloc(R * Hdim);
  float* o1   = alloc((size_t)Ndim * Hdim);

  // ushort region (bf16 activations + weights) after the fp32 region
  unsigned short* wsb = (unsigned short*)(ws + off);
  size_t uoff = 0;
  auto balloc = [&](size_t n) { unsigned short* p = wsb + uoff; uoff += n; return p; };
  // bf16 activation mirrors
  unsigned short* hn_bf   = balloc(R * Hdim);   // also fe1 output
  unsigned short* att_bf  = balloc(R * Hdim);
  unsigned short* abuf_bf = balloc(R * Hdim);
  unsigned short* ht_bf   = balloc(R * Hdim);
  unsigned short* ff1_bf  = balloc(R * 4 * Hdim);
  unsigned short* ea_bf   = balloc((size_t)Sdim * Edim * 3);
  // bf16 weight copies
  unsigned short* fe_w2b  = balloc((size_t)Hdim * Hdim);
  unsigned short* g_inwb  = balloc((size_t)Ldim * 3 * Hdim * Hdim);
  unsigned short* g_outwb = balloc((size_t)Ldim * Hdim * Hdim);
  unsigned short* g_m1wb  = balloc((size_t)Ldim * Hdim * (Hdim + 32));  // K padded to 288
  unsigned short* g_m2wb  = balloc((size_t)Ldim * Hdim * Hdim);
  unsigned short* t_inwb  = balloc((size_t)Ldim * 3 * Hdim * Hdim);
  unsigned short* t_outwb = balloc((size_t)Ldim * Hdim * Hdim);
  unsigned short* t_l1wb  = balloc((size_t)Ldim * 4 * Hdim * Hdim);
  unsigned short* t_l2wb  = balloc((size_t)Ldim * Hdim * 4 * Hdim);
  unsigned short* o_w1b   = balloc((size_t)Hdim * Hdim);

  const dim3 blk128(128), blk256(256);
  const unsigned Ru = (unsigned)R;

  // ---- one-time (per launch) weight conversion to bf16 ----
  f32_to_bf16_kernel<<<256, blk256, 0, stream>>>(fe_w2, fe_w2b, (long)Hdim * Hdim);
  f32_to_bf16_kernel<<<1024, blk256, 0, stream>>>(g_inw, g_inwb, (long)Ldim * 3 * Hdim * Hdim);
  f32_to_bf16_kernel<<<512, blk256, 0, stream>>>(g_outw, g_outwb, (long)Ldim * Hdim * Hdim);
  f32_to_bf16_pad_kernel<<<((unsigned)(Ldim * Hdim * (Hdim + 32)) + 255) / 256, blk256, 0,
                           stream>>>(g_m1w, g_m1wb, Ldim * Hdim, Hdim + 3, Hdim + 32);
  f32_to_bf16_kernel<<<512, blk256, 0, stream>>>(g_m2w, g_m2wb, (long)Ldim * Hdim * Hdim);
  f32_to_bf16_kernel<<<1024, blk256, 0, stream>>>(t_inw, t_inwb, (long)Ldim * 3 * Hdim * Hdim);
  f32_to_bf16_kernel<<<512, blk256, 0, stream>>>(t_outw, t_outwb, (long)Ldim * Hdim * Hdim);
  f32_to_bf16_kernel<<<1024, blk256, 0, stream>>>(t_l1w, t_l1wb, (long)Ldim * 4 * Hdim * Hdim);
  f32_to_bf16_kernel<<<1024, blk256, 0, stream>>>(t_l2w, t_l2wb, (long)Ldim * Hdim * 4 * Hdim);
  f32_to_bf16_kernel<<<256, blk256, 0, stream>>>(o_w1, o_w1b, (long)Hdim * Hdim);

  // ---- feature encoder ----
  fe1_kernel<<<(Ru * Hdim + 255) / 256, blk256, 0, stream>>>(x, fe_w1, fe_b1,
                                                             hn_bf, (int)R);
  gemm_bias_kernel<0, 0, 0><<<dim3(Ru / 64, Hdim / 64), blk128, 0, stream>>>(
      hn_bf, fe_w2b, fe_b2, nullptr, h, nullptr, (int)R, Hdim, Hdim, Hdim,
      Hdim, Hdim);
  ea_kernel<<<(Sdim * Edim + 255) / 256, blk256, 0, stream>>>(h, rowi, coli,
                                                              ea_bf);
  zero_kernel<<<4, blk256, 0, stream>>>(cnt, (long)Ndim);
  count_kernel<<<(Edim + 255) / 256, blk256, 0, stream>>>(rowi, cnt);
  inv_kernel<<<4, blk256, 0, stream>>>(cnt, Ndim);

  // ---- graph layers ----
  for (int l = 0; l < Ldim; ++l) {
    ln_kernel<<<Ru, blk256, 0, stream>>>(h, nullptr, nullptr, 1,
                                         g_n1g + l * Hdim, g_n1b + l * Hdim,
                                         nullptr, hn, hn_bf);
    gemm_bias_kernel<0, 0, 0>
        <<<dim3(Ru / 64, (3 * Hdim) / 64), blk128, 0, stream>>>(
            hn_bf, g_inwb + (size_t)l * 3 * Hdim * Hdim, g_inb + l * 3 * Hdim,
            nullptr, qkv, nullptr, (int)R, 3 * Hdim, Hdim, Hdim, Hdim,
            3 * Hdim);
    graph_attn_kernel<<<(Sdim * NHdim * Ndim) / 8, blk256, 0, stream>>>(
        qkv, att_bf);
    gemm_bias_kernel<0, 1, 1><<<dim3(Ru / 64, Hdim / 64), blk128, 0, stream>>>(
        att_bf, g_outwb + (size_t)l * Hdim * Hdim, g_outb + l * Hdim, h, abuf,
        abuf_bf, (int)R, Hdim, Hdim, Hdim, Hdim, Hdim);
    zero_kernel<<<2048, blk256, 0, stream>>>(agg, (long)R * Hdim);
    edge_mlp_kernel<<<(Sdim * Edim) / 64, blk128, 0, stream>>>(
        abuf_bf, ea_bf, rowi, coli, g_m1wb + (size_t)l * Hdim * (Hdim + 32),
        g_m1b + l * Hdim, g_m2wb + (size_t)l * Hdim * Hdim, g_m2b + l * Hdim,
        agg);
    ln_kernel<<<Ru, blk256, 0, stream>>>(agg, nullptr, cnt, Ndim,
                                         g_n2g + l * Hdim, g_n2b + l * Hdim, h,
                                         h, nullptr);
  }

  // ---- transpose (S,N,H) -> (N,S,H) ----
  transpose_kernel<<<(Ru * Hdim + 255) / 256, blk256, 0, stream>>>(h, ht,
                                                                   ht_bf);

  // ---- transformer layers over T=2 ----
  for (int l = 0; l < Ldim; ++l) {
    gemm_bias_kernel<0, 0, 0>
        <<<dim3(Ru / 64, (3 * Hdim) / 64), blk128, 0, stream>>>(
            ht_bf, t_inwb + (size_t)l * 3 * Hdim * Hdim, t_inb + l * 3 * Hdim,
            nullptr, qkv, nullptr, (int)R, 3 * Hdim, Hdim, Hdim, Hdim,
            3 * Hdim);
    t_attn_kernel<<<(Ndim * NHdim + 255) / 256, blk256, 0, stream>>>(qkv,
                                                                     att_bf);
    gemm_bias_kernel<0, 0, 0><<<dim3(Ru / 64, Hdim / 64), blk128, 0, stream>>>(
        att_bf, t_outwb + (size_t)l * Hdim * Hdim, t_outb + l * Hdim, nullptr,
        abuf, nullptr, (int)R, Hdim, Hdim, Hdim, Hdim, Hdim);
    ln_kernel<<<Ru, blk256, 0, stream>>>(ht, abuf, nullptr, 1,
                                         t_n1g + l * Hdim, t_n1b + l * Hdim,
                                         nullptr, ht, ht_bf);
    gemm_bias_kernel<1, 0, 1>
        <<<dim3(Ru / 64, (4 * Hdim) / 64), blk128, 0, stream>>>(
            ht_bf, t_l1wb + (size_t)l * 4 * Hdim * Hdim, t_l1b + l * 4 * Hdim,
            nullptr, ff1, ff1_bf, (int)R, 4 * Hdim, Hdim, Hdim, Hdim,
            4 * Hdim);
    gemm_bias_kernel<0, 0, 0><<<dim3(Ru / 64, Hdim / 64), blk128, 0, stream>>>(
        ff1_bf, t_l2wb + (size_t)l * Hdim * 4 * Hdim, t_l2b + l * Hdim,
        nullptr, ff2, nullptr, (int)R, Hdim, 4 * Hdim, 4 * Hdim, 4 * Hdim,
        Hdim);
    ln_kernel<<<Ru, blk256, 0, stream>>>(ht, ff2, nullptr, 1,
                                         t_n2g + l * Hdim, t_n2b + l * Hdim,
                                         nullptr, ht, ht_bf);
  }

  // ---- head: last token (s=1) -> relu(o_w1) -> o_w2 ----
  gemm_bias_kernel<1, 0, 0><<<dim3(Ndim / 64, Hdim / 64), blk128, 0, stream>>>(
      ht_bf + Hdim, o_w1b, o_b1, nullptr, o1, nullptr, Ndim, Hdim, Hdim,
      2 * Hdim, Hdim, Hdim);
  out_kernel<<<(Ndim * 3 + 255) / 256, blk256, 0, stream>>>(o1, o_w2, o_b2,
                                                            (float*)d_out);
}
